// QRNNLayer_84774064488818
// MI455X (gfx1250) — compile-verified, matrix operand order checked by reading, not run on hardware
//
#include <hip/hip_runtime.h>
#include <hip/hip_bf16.h>
#include <math.h>
#include <stdint.h>

// ---------------------------------------------------------------------------
// QRNN forward for MI455X (gfx1250), wave32 + WMMA bf16.
//   Phase 1: X and W (f32) -> bf16 in workspace.
//   Phase 2: GEMM  Y = X @ W^T + b, fused activations -> gate planes (f32).
//            256x128 block tile, double-buffered LDS filled with
//            GLOBAL_LOAD_ASYNC_TO_LDS_B128 (ASYNCcnt), one barrier per chunk.
//   Phase 3: sequential fo-pool scan + output gate.
// ---------------------------------------------------------------------------

typedef __attribute__((ext_vector_type(16))) __bf16 v16bf;
typedef __attribute__((ext_vector_type(8)))  __bf16 v8bf;
typedef __attribute__((ext_vector_type(4)))  __bf16 v4bf;
typedef __attribute__((ext_vector_type(8)))  float  v8f;
typedef __attribute__((ext_vector_type(4)))  float  v4f;

#define S_DIM   2048
#define B_DIM   16
#define D_DIM   1024
#define H_DIM   1024
#define M_DIM   (S_DIM * B_DIM)       // 32768 rows
#define N_DIM   (3 * H_DIM)           // 3072 cols

#define MBLK    256                   // block M tile
#define NBLK    128                   // block N tile
#define KC      32                    // K chunk (one bf16 WMMA K)
#define NK      (D_DIM / KC)          // 32 chunks
#define NT      8                     // N tiles (16 wide) per wave
#define MT      2                     // M tiles (16 tall) per wave

// LDS row stride: 64B of data + 16B pad -> stride 20 dwords. l16*20 mod 64 is a
// permutation of 16 distinct bank quads -> conflict-free ds_load_b128 fragments.
#define A_STRIDE 80
#define B_STRIDE 80
#define A_LDS_BYTES (MBLK * A_STRIDE)            // 20480
#define B_LDS_BYTES (NBLK * B_STRIDE)            // 10240
#define BUF_BYTES   (A_LDS_BYTES + B_LDS_BYTES)  // 30720 per buffer, x2 buffers

// ---- CDNA5 async memory->LDS (ASYNCcnt) via inline asm (toolchain-portable) ----
__device__ __forceinline__ void async_ld_b128(unsigned ldsAddr, const void* g) {
    asm volatile("global_load_async_to_lds_b128 %0, %1, off"
                 :: "v"(ldsAddr), "v"((uint64_t)(uintptr_t)g)
                 : "memory");
}
__device__ __forceinline__ void wait_async0() {
    asm volatile("s_wait_asynccnt 0x0" ::: "memory");
}

// ---------------------------------------------------------------------------
// Kernel 1: vectorized f32 -> bf16 (4 elems/thread). Used for both X and W.
// ---------------------------------------------------------------------------
__global__ void cvt_f32_bf16(const float* __restrict__ src, __bf16* __restrict__ dst, int n4) {
    int i = blockIdx.x * blockDim.x + threadIdx.x;
    if (i < n4) {
        v4f v = *(const v4f*)(src + (size_t)i * 4);
        v4bf o;
        o[0] = (__bf16)v[0]; o[1] = (__bf16)v[1];
        o[2] = (__bf16)v[2]; o[3] = (__bf16)v[3];
        *(v4bf*)(dst + (size_t)i * 4) = o;
    }
}

// ---------------------------------------------------------------------------
// Kernel 2: async-LDS double-buffered WMMA GEMM + bias + activation.
// Block: 256 threads = 8 waves; tile 256(M) x 128(N); grid = (128, 24).
// Wave w computes rows [blk*256 + w*32, +32), all 128 cols of the strip.
//
// A fragment (16-bit A 16x32, wave32): lane L (row M = L&15, half = L>>4):
//   elems 0..7  = K kBase +      half*8 .. +7   -> LDS chunk (half)
//   elems 8..15 = K kBase + 16 + half*8 .. +7   -> LDS chunk (2+half)
// B mirrors A with lane -> column N.
// ---------------------------------------------------------------------------
__global__ void __launch_bounds__(256)
qrnn_gemm_gates(const __bf16* __restrict__ Xb,
                const __bf16* __restrict__ Wb,
                const float* __restrict__ bias,
                float* __restrict__ gZ,
                float* __restrict__ gF,
                float* __restrict__ gO) {
    __shared__ __align__(16) char smem[2 * BUF_BYTES];

    const int tid   = threadIdx.x;
    const int lane  = tid & 31;
    const int wave  = tid >> 5;
    const int half  = lane >> 4;
    const int l16   = lane & 15;
    const int mBlk  = blockIdx.x * MBLK;
    const int nBase = blockIdx.y * NBLK;
    const int mW    = wave * (MT * 16);       // wave's 32-row window in the tile

    // Staging sources: thread i stages A row i (4xB128); 2 threads per B column
    // (2xB128 each) so all 256 threads participate -> no exec-mask churn.
    const __bf16* aSrc = Xb + (size_t)(mBlk + tid) * D_DIM;
    const __bf16* bSrc = Wb + (size_t)(nBase + (tid >> 1)) * D_DIM + (tid & 1) * 16;

    // Per-thread LDS destination offsets (wave-relative LDS byte addresses).
    const unsigned ldsBase = (unsigned)(uintptr_t)(void*)smem;
    const unsigned aDst    = ldsBase + (unsigned)(tid * A_STRIDE);
    const unsigned bDst    = ldsBase + (unsigned)(A_LDS_BYTES + (tid >> 1) * B_STRIDE + (tid & 1) * 32);

    // Prologue: issue async fill of buffer 0 with chunk 0.
#pragma unroll
    for (int c = 0; c < 4; ++c) async_ld_b128(aDst + c * 16, aSrc + c * 8);
    async_ld_b128(bDst,      bSrc);
    async_ld_b128(bDst + 16, bSrc + 8);

    v8f acc[MT][NT] = {};

    for (int kc = 0; kc < NK; ++kc) {
        const unsigned cur = (kc & 1) ? BUF_BYTES : 0;

        wait_async0();                         // my fills of buf[cur] complete
        __syncthreads();                       // all fills visible; buf[next] free

        // Refill the other buffer with chunk kc+1 (overlaps WMMAs below).
        if (kc + 1 < NK) {
            const unsigned nxt  = ((kc + 1) & 1) ? BUF_BYTES : 0;
            const int      kOff = (kc + 1) * KC;
#pragma unroll
            for (int c = 0; c < 4; ++c)
                async_ld_b128(aDst + nxt + c * 16, aSrc + kOff + c * 8);
            async_ld_b128(bDst + nxt,      bSrc + kOff);
            async_ld_b128(bDst + nxt + 16, bSrc + kOff + 8);
        }

        const char* aS = smem + cur;
        const char* bS = smem + cur + A_LDS_BYTES;
        const int c0 = half * 16;              // byte offset of first chunk
        const int c1 = (2 + half) * 16;        // byte offset of second chunk

        v16bf af[MT];
#pragma unroll
        for (int mt = 0; mt < MT; ++mt) {
            const char* arow = aS + (mW + mt * 16 + l16) * A_STRIDE;
            v8bf lo = *(const v8bf*)(arow + c0);
            v8bf hi = *(const v8bf*)(arow + c1);
            af[mt] = __builtin_shufflevector(lo, hi, 0, 1, 2, 3, 4, 5, 6, 7,
                                             8, 9, 10, 11, 12, 13, 14, 15);
        }

#pragma unroll
        for (int t = 0; t < NT; ++t) {
            const char* brow = bS + (t * 16 + l16) * B_STRIDE;
            v8bf lo = *(const v8bf*)(brow + c0);
            v8bf hi = *(const v8bf*)(brow + c1);
            v16bf bf = __builtin_shufflevector(lo, hi, 0, 1, 2, 3, 4, 5, 6, 7,
                                               8, 9, 10, 11, 12, 13, 14, 15);
            acc[0][t] = __builtin_amdgcn_wmma_f32_16x16x32_bf16(
                false, af[0], false, bf, (short)0, acc[0][t], false, false);
            acc[1][t] = __builtin_amdgcn_wmma_f32_16x16x32_bf16(
                false, af[1], false, bf, (short)0, acc[1][t], false, false);
        }
    }

    // ---- epilogue: bias + activation, store to the gate plane ----
    // The 128-wide strip lies entirely in one gate region (1024 % 128 == 0).
    const int gate = nBase >> 10;              // 0 = Z (tanh), 1 = F, 2 = O (sigmoid)
    float* gptr = (gate == 0) ? gZ : ((gate == 1) ? gF : gO);

#pragma unroll
    for (int mt = 0; mt < MT; ++mt) {
#pragma unroll
        for (int t = 0; t < NT; ++t) {
            const int n    = nBase + t * 16 + l16;
            const int ncol = n & (H_DIM - 1);
            const float bv = bias[n];
#pragma unroll
            for (int r = 0; r < 8; ++r) {
                const int row = mBlk + mW + mt * 16 + r + half * 8;
                const float v = acc[mt][t][r] + bv;
                float res;
                if (gate == 0) {
                    // tanh(v) = 2*sigmoid(2v) - 1  (saturates correctly)
                    res = 2.0f / (1.0f + __expf(-2.0f * v)) - 1.0f;
                } else {
                    res = 1.0f / (1.0f + __expf(-v));
                }
                gptr[(size_t)row * H_DIM + ncol] = res;
            }
        }
    }
}

// ---------------------------------------------------------------------------
// Kernel 3: fo-pool scan. One thread per (b,h) channel (16384 channels).
//   c_s = f*z + (1-f)*c = fma(f, z - c, c);  out = o * c; last c appended.
// ---------------------------------------------------------------------------
__global__ void __launch_bounds__(256)
qrnn_scan(const float* __restrict__ gZ,
          const float* __restrict__ gF,
          const float* __restrict__ gO,
          const float* __restrict__ hidden,
          float* __restrict__ out) {
    const int t = blockIdx.x * blockDim.x + threadIdx.x;   // 0..16383
    const int stride = B_DIM * H_DIM;                      // 16384
    float c = hidden[t];
#pragma unroll 4
    for (int s = 0; s < S_DIM; ++s) {
        const size_t idx = (size_t)s * stride + t;
        const float z = gZ[idx];
        const float f = gF[idx];
        const float o = gO[idx];
        c = fmaf(f, z - c, c);
        out[idx] = o * c;
    }
    out[(size_t)S_DIM * stride + t] = c;                   // final cell state
}

// ---------------------------------------------------------------------------
// Launch. Inputs: d_in[0]=X [S,B,D] f32, d_in[1]=hidden [B,H] f32,
//                 d_in[2]=W [3H,D] f32, d_in[3]=b [3H] f32.
// d_out: H_out [S,B,H] f32 followed by C_last [1,B,H] f32.
// Workspace: [0,6MB) W_bf16; [8MB,72MB) X_bf16; [80MB,80MB+384MB) gates f32.
// ---------------------------------------------------------------------------
extern "C" void kernel_launch(void* const* d_in, const int* in_sizes, int n_in,
                              void* d_out, int out_size, void* d_ws, size_t ws_size,
                              hipStream_t stream) {
    const float* X      = (const float*)d_in[0];
    const float* hidden = (const float*)d_in[1];
    const float* W      = (const float*)d_in[2];
    const float* bias   = (const float*)d_in[3];
    float* out          = (float*)d_out;

    char* ws = (char*)d_ws;
    __bf16* Wb = (__bf16*)ws;                                   // 6 MiB
    __bf16* Xb = (__bf16*)(ws + (size_t)8 * 1024 * 1024);       // 64 MiB
    const size_t gateElems = (size_t)M_DIM * H_DIM;             // 33,554,432
    float* gZ = (float*)(ws + (size_t)80 * 1024 * 1024);
    float* gF = gZ + gateElems;
    float* gO = gF + gateElems;

    // Phase 1: f32 -> bf16 for W and X.
    {
        const int nW4 = (N_DIM * D_DIM) / 4;                    // 786,432
        cvt_f32_bf16<<<(nW4 + 255) / 256, 256, 0, stream>>>(W, Wb, nW4);
        const int nX4 = (M_DIM * D_DIM) / 4;                    // 8,388,608
        cvt_f32_bf16<<<(nX4 + 255) / 256, 256, 0, stream>>>(X, Xb, nX4);
    }

    // Phase 2: async-LDS double-buffered GEMM + fused activations.
    {
        dim3 grid(M_DIM / MBLK, N_DIM / NBLK);                  // (128, 24)
        qrnn_gemm_gates<<<grid, 256, 0, stream>>>(Xb, Wb, bias, gZ, gF, gO);
    }

    // Phase 3: recurrence + output gate.
    {
        const int channels = B_DIM * H_DIM;                     // 16384
        qrnn_scan<<<channels / 256, 256, 0, stream>>>(gZ, gF, gO, hidden, out);
    }
}
